// IntraDistanceLoss1_59717225283876
// MI455X (gfx1250) — matrix operand order repaired and checked by reading, not compiled, not run on hardware
//
#include <hip/hip_runtime.h>

typedef __bf16 bf16_t;
typedef __attribute__((ext_vector_type(16))) __bf16 v16bf;
typedef __attribute__((ext_vector_type(8)))  __bf16 v8bf;
typedef __attribute__((ext_vector_type(8)))  float  v8f;

#define DDIM 256
#define CREAL 500
#define CPAD 512
#define LAMBDA_INTRA 0.1f
#define EPS_D2 1e-12f

// ---------- bf16 bit helpers (truncation split: x = hi + lo, ~16-bit mantissa total) ----------
static __device__ __forceinline__ unsigned short f32_to_bf16_bits(float x) {
  return (unsigned short)(__float_as_uint(x) >> 16);
}
static __device__ __forceinline__ float bf16_bits_to_f32(unsigned short h) {
  return __uint_as_float(((unsigned)h) << 16);
}
static __device__ __forceinline__ bf16_t bits_to_bf16(unsigned short h) {
  bf16_t r; __builtin_memcpy(&r, &h, sizeof(r)); return r;
}

// ---------- pre-pass: centers f32 -> zero-padded [512][256] bf16 hi/lo planes ----------
__global__ __launch_bounds__(256) void cvt_centers_kernel(
    const float* __restrict__ centers,
    unsigned short* __restrict__ bhi, unsigned short* __restrict__ blo) {
  int idx = blockIdx.x * 256 + threadIdx.x;          // [0, 512*256)
  float f = (idx < CREAL * DDIM) ? centers[idx] : 0.0f;
  unsigned short h = f32_to_bf16_bits(f);
  float resid = f - bf16_bits_to_f32(h);
  bhi[idx] = h;
  blo[idx] = f32_to_bf16_bits(resid);
}

// ---------- pre-pass: c2[c] = ||center_c||^2, padded with zeros ----------
__global__ __launch_bounds__(256) void center_norms_kernel(
    const float* __restrict__ centers, float* __restrict__ c2) {
  int c = blockIdx.x * 256 + threadIdx.x;
  if (c >= CPAD) return;
  float s = 0.0f;
  if (c < CREAL) {
    const float* row = centers + (size_t)c * DDIM;
    for (int k = 0; k < DDIM; ++k) { float v = row[k]; s += v * v; }
  }
  c2[c] = s;
}

// ---------- main: one wave32 per 16-row strip; split-bf16 WMMA GEMM + fused dist epilogue ----------
__global__ __launch_bounds__(256) void strip_kernel(
    const float* __restrict__ feat,
    const int* __restrict__ labels,
    const float* __restrict__ centers,
    const unsigned short* __restrict__ bhi,
    const unsigned short* __restrict__ blo,
    const float* __restrict__ c2g,
    float* __restrict__ partial) {
  const int lane  = threadIdx.x & 31;
  const int wave  = threadIdx.x >> 5;
  const int strip = blockIdx.x * 8 + wave;
  const int Mbase = strip * 16;
  const int half  = lane >> 4;   // 0: lanes 0-15, 1: lanes 16-31
  const int l15   = lane & 15;

  // --- Load A fragments for full K=256 (ISA 16-bit A 16x32 layout: per lane two
  // contiguous 8-element K chunks at half*8 and 16+half*8), split into bf16 hi/lo.
  v16bf a_hi[8], a_lo[8];
  float x2p = 0.0f;
  const float* arow = feat + (size_t)(Mbase + l15) * DDIM;
  #pragma unroll
  for (int k8 = 0; k8 < 8; ++k8) {
    const int kb = k8 * 32 + half * 8;
    v8f f0 = *(const v8f*)(arow + kb);        // K chunk [kb, kb+8)
    v8f f1 = *(const v8f*)(arow + kb + 16);   // K chunk [kb+16, kb+24)
    v16bf hi, lo;
    #pragma unroll
    for (int j = 0; j < 8; ++j) {
      float x = f0[j];
      unsigned short hb = f32_to_bf16_bits(x);
      hi[j]     = bits_to_bf16(hb);
      lo[j]     = bits_to_bf16(f32_to_bf16_bits(x - bf16_bits_to_f32(hb)));
      x2p += x * x;
      float y = f1[j];
      unsigned short hb2 = f32_to_bf16_bits(y);
      hi[8 + j] = bits_to_bf16(hb2);
      lo[8 + j] = bits_to_bf16(f32_to_bf16_bits(y - bf16_bits_to_f32(hb2)));
      x2p += y * y;
    }
    a_hi[k8] = hi;
    a_lo[k8] = lo;
  }
  // Row |x|^2: lanes L and L+16 each hold half of row Mbase+l15.
  float x2row = x2p + __shfl_xor(x2p, 16, 32);

  // Redistribute x2 + labels to match C/D layout rows: lane covers rows Mbase + half*8 + r.
  float x2r[8];
  int lab[8];
  #pragma unroll
  for (int r = 0; r < 8; ++r) {
    x2r[r] = __shfl(x2row, half * 8 + r, 32);
    lab[r] = labels[Mbase + half * 8 + r];
  }

  float rowsum[8], down[8];
  #pragma unroll
  for (int r = 0; r < 8; ++r) { rowsum[r] = 0.0f; down[r] = 0.0f; }

  // --- Loop over 32 N-tiles of 16 centers (padded to 512).
  for (int nt = 0; nt < 32; ++nt) {
    const int c = nt * 16 + l15;   // this lane's output column
    // B 32x16 layout: lane holds column c, two 8-element K chunks at half*8 and 16+half*8.
    const unsigned short* bH = bhi + (size_t)c * DDIM + half * 8;
    const unsigned short* bL = blo + (size_t)c * DDIM + half * 8;
    v8f acc0 = {}, acc1 = {}, acc2 = {};   // hi*hi, lo*hi, hi*lo — 3 independent chains
    #pragma unroll
    for (int k8 = 0; k8 < 8; ++k8) {
      v16bf bh = __builtin_shufflevector(*(const v8bf*)(bH + k8 * 32), *(const v8bf*)(bH + k8 * 32 + 16), 0,1,2,3,4,5,6,7,8,9,10,11,12,13,14,15);
      v16bf bl = __builtin_shufflevector(*(const v8bf*)(bL + k8 * 32), *(const v8bf*)(bL + k8 * 32 + 16), 0,1,2,3,4,5,6,7,8,9,10,11,12,13,14,15);
      acc0 = __builtin_amdgcn_wmma_f32_16x16x32_bf16(false, a_hi[k8], false, bh,
                                                     (short)0, acc0, false, false);
      acc1 = __builtin_amdgcn_wmma_f32_16x16x32_bf16(false, a_lo[k8], false, bh,
                                                     (short)0, acc1, false, false);
      acc2 = __builtin_amdgcn_wmma_f32_16x16x32_bf16(false, a_hi[k8], false, bl,
                                                     (short)0, acc2, false, false);
    }
    const float cc2 = c2g[c];
    const bool inr = (c < CREAL);
    #pragma unroll
    for (int r = 0; r < 8; ++r) {
      float dot  = acc0[r] + acc1[r] + acc2[r];
      float d2   = x2r[r] + cc2 - 2.0f * dot;
      float dist = __builtin_sqrtf(fmaxf(d2, EPS_D2));
      rowsum[r] += inr ? dist : 0.0f;
      down[r]    = (inr && (c == lab[r])) ? dist : down[r];
    }
  }

  // --- Reduce dist sums across the 16 lanes of each half (butterfly).
  #pragma unroll
  for (int r = 0; r < 8; ++r) {
    #pragma unroll
    for (int m = 1; m <= 8; m <<= 1) {
      rowsum[r] += __shfl_xor(rowsum[r], m, 32);
      down[r]   += __shfl_xor(down[r], m, 32);
    }
  }

  // --- Exact own-class norms (fp32): whole wave per row, 8 elems per lane.
  float cor[16];
  #pragma unroll
  for (int rr = 0; rr < 16; ++rr) {
    const int m = Mbase + rr;
    const int lb = labels[m];
    v8f fv = *(const v8f*)(feat    + (size_t)m  * DDIM + lane * 8);
    v8f cv = *(const v8f*)(centers + (size_t)lb * DDIM + lane * 8);
    float s = 0.0f;
    #pragma unroll
    for (int j = 0; j < 8; ++j) { float d = fv[j] - cv[j]; s += d * d; }
    #pragma unroll
    for (int mm = 1; mm <= 16; mm <<= 1) s += __shfl_xor(s, mm, 32);
    cor[rr] = __builtin_sqrtf(s);   // identical in all lanes
  }

  // --- Per-strip contribution: sum over 16 rows of correct/incorrect + 0.1*correct.
  float psum = 0.0f;
  #pragma unroll
  for (int r = 0; r < 8; ++r) {
    float inc = (rowsum[r] - down[r]) * (1.0f / (float)(CREAL - 1));
    float cr  = half ? cor[8 + r] : cor[r];
    psum += cr / inc + LAMBDA_INTRA * cr;
  }
  psum += __shfl_xor(psum, 16, 32);
  if (lane == 0) partial[strip] = psum;
}

// ---------- deterministic final reduction ----------
__global__ __launch_bounds__(256) void reduce_kernel(
    const float* __restrict__ partial, float* __restrict__ out, int n, float inv_b) {
  __shared__ float sm[256];
  float s = 0.0f;
  for (int i = threadIdx.x; i < n; i += 256) s += partial[i];
  sm[threadIdx.x] = s;
  __syncthreads();
  for (int off = 128; off > 0; off >>= 1) {
    if (threadIdx.x < off) sm[threadIdx.x] += sm[threadIdx.x + off];
    __syncthreads();
  }
  if (threadIdx.x == 0) out[0] = sm[0] * inv_b;
}

extern "C" void kernel_launch(void* const* d_in, const int* in_sizes, int n_in,
                              void* d_out, int out_size, void* d_ws, size_t ws_size,
                              hipStream_t stream) {
  const float* feat    = (const float*)d_in[0];
  const int*   labels  = (const int*)d_in[1];
  const float* centers = (const float*)d_in[2];

  const int B       = in_sizes[0] / DDIM;   // 131072
  const int nstrips = B / 16;               // 8192
  const int nblocks = nstrips / 8;          // 1024 (8 waves / block)

  // Workspace layout (all 256B-aligned chunks):
  char* ws = (char*)d_ws;
  unsigned short* bhi = (unsigned short*)ws;                        // 512*256*2 = 262144 B
  unsigned short* blo = (unsigned short*)(ws + 262144);             // 262144 B
  float* c2           = (float*)(ws + 524288);                      // 512*4 = 2048 B
  float* partial      = (float*)(ws + 526336);                      // nstrips*4 B
  float* out          = (float*)d_out;

  cvt_centers_kernel<<<CPAD, 256, 0, stream>>>(centers, bhi, blo);
  center_norms_kernel<<<CPAD / 256, 256, 0, stream>>>(centers, c2);
  strip_kernel<<<nblocks, 256, 0, stream>>>(feat, labels, centers, bhi, blo, c2, partial);
  reduce_kernel<<<1, 256, 0, stream>>>(partial, out, nstrips, 1.0f / (float)B);
}
